// ParallelAttention_router_33629593928080
// MI455X (gfx1250) — matrix operand order repaired and checked by reading, not compiled, not run on hardware
//
#include <hip/hip_runtime.h>

typedef __attribute__((ext_vector_type(16))) __bf16        v16bf;
typedef __attribute__((ext_vector_type(8)))  float         v8f;
typedef __attribute__((ext_vector_type(8)))  unsigned int  v8u;

#define H            2048
#define Z            64
#define NTOK         16384
#define KSTEPS       (H / 32)      // 64 WMMA K-steps
#define CHUNKS_PER_W 8192          // 64 kk * 4 nt * 32 lanes
#define TOK_PER_WG   64
#define LOG2E        1.4426950408889634f

// fp32 -> bf16, round-to-nearest-even
__device__ __forceinline__ unsigned int f2bf(float f) {
    unsigned int u = __float_as_uint(f);
    return (u + 0x7FFFu + ((u >> 16) & 1u)) >> 16;
}

// ---------------------------------------------------------------------------
// Kernel 1: convert W (h x z, fp32 row-major) into bf16 chunks pre-swizzled
// into a contiguous-K B-fragment order for V_WMMA_F32_16X16X32_BF16 (same K order as A):
//   lanes 0-15 : N = lane,    K = kk*32 + 0..15   (pairs packed per dword)
//   lanes 16-31: N = lane-16, K = kk*32 + 16..31
// Chunk c = ((kk*4 + nt)*32 + lane); each chunk = 16 bf16 = 32 bytes.
// ---------------------------------------------------------------------------
__global__ void swizzle_weights(const float* __restrict__ Wq,
                                const float* __restrict__ Wk,
                                const float* __restrict__ Wv,
                                unsigned int* __restrict__ wsw) {
    int cid  = blockIdx.x * blockDim.x + threadIdx.x;   // 0 .. 3*8192-1
    int w    = cid >> 13;
    int r    = cid & (CHUNKS_PER_W - 1);
    int kk   = r >> 7;
    int nt   = (r >> 5) & 3;
    int lane = r & 31;
    int n    = nt * 16 + (lane & 15);
    int kb   = kk * 32 + ((lane >> 4) << 4);
    const float* W = (w == 0) ? Wq : (w == 1) ? Wk : Wv;
    unsigned int* dst = wsw + (size_t)cid * 8;
    #pragma unroll
    for (int j = 0; j < 8; ++j) {
        float f0 = W[(size_t)(kb + 2 * j)     * Z + n];
        float f1 = W[(size_t)(kb + 2 * j + 1) * Z + n];
        dst[j] = f2bf(f0) | (f2bf(f1) << 16);
    }
}

// ---------------------------------------------------------------------------
// Kernel 2: fused  q,k,v = x@W{q,k,v}  (bf16 WMMA, f32 accumulate)
//           + per-token softmax(q_i * k_j) @ v
// 128 threads = 4 waves; each wave does a 16-token M-tile; WG covers 64 tokens.
// ---------------------------------------------------------------------------
__global__ void __launch_bounds__(128)
qkv_attn_fused(const float* __restrict__ x,
               const unsigned int* __restrict__ wsw,
               float* __restrict__ out) {
    __shared__ float sq[TOK_PER_WG][Z];
    __shared__ float sk[TOK_PER_WG][Z];
    __shared__ float sv[TOK_PER_WG][Z];

    const int tid  = threadIdx.x;
    const int lane = tid & 31;
    const int wave = tid >> 5;
    const int lmod = lane & 15;
    const int lhi  = lane >> 4;                       // 0 or 1

    const int tokBase = blockIdx.x * TOK_PER_WG;
    const int row     = tokBase + wave * 16 + lmod;   // A-fragment row (token)

    v8f acc[3][4];
    #pragma unroll
    for (int w = 0; w < 3; ++w)
        #pragma unroll
        for (int nt = 0; nt < 4; ++nt)
            acc[w][nt] = {};

    const float* arow  = x + (size_t)row * H;
    const v8u* chunks  = (const v8u*)wsw;             // 32-byte B chunks

    for (int kk = 0; kk < KSTEPS; ++kk) {
        // ---- A fragment (16x32 bf16): lanes 0-15 K{0..15}, 16-31 K{16..31}
        const float* ap = arow + kk * 32;
        const int off0  = lhi * 16;
        float4 r0a = *(const float4*)(ap + off0);
        float4 r0b = *(const float4*)(ap + off0 + 4);
        float4 r1a = *(const float4*)(ap + off0 + 8);
        float4 r1b = *(const float4*)(ap + off0 + 12);
        v8u au;
        au[0] = f2bf(r0a.x) | (f2bf(r0a.y) << 16);
        au[1] = f2bf(r0a.z) | (f2bf(r0a.w) << 16);
        au[2] = f2bf(r0b.x) | (f2bf(r0b.y) << 16);
        au[3] = f2bf(r0b.z) | (f2bf(r0b.w) << 16);
        au[4] = f2bf(r1a.x) | (f2bf(r1a.y) << 16);
        au[5] = f2bf(r1a.z) | (f2bf(r1a.w) << 16);
        au[6] = f2bf(r1b.x) | (f2bf(r1b.y) << 16);
        au[7] = f2bf(r1b.z) | (f2bf(r1b.w) << 16);
        v16bf a = __builtin_bit_cast(v16bf, au);

        #pragma unroll
        for (int w = 0; w < 3; ++w) {
            #pragma unroll
            for (int nt = 0; nt < 4; ++nt) {
                v8u bu = chunks[(size_t)w * CHUNKS_PER_W + kk * 128 + nt * 32 + lane];
                v16bf b = __builtin_bit_cast(v16bf, bu);
                acc[w][nt] = __builtin_amdgcn_wmma_f32_16x16x32_bf16(
                    false, a, false, b, (short)0, acc[w][nt], false, false);
            }
        }
    }

    // ---- spill q,k,v fragments to LDS (C/D layout: VGPR r -> M = r + lhi*8, N = lmod)
    #pragma unroll
    for (int w = 0; w < 3; ++w) {
        float* dst = (w == 0) ? &sq[0][0] : (w == 1) ? &sk[0][0] : &sv[0][0];
        #pragma unroll
        for (int nt = 0; nt < 4; ++nt) {
            #pragma unroll
            for (int r = 0; r < 8; ++r) {
                int tok = wave * 16 + r + lhi * 8;
                dst[tok * Z + nt * 16 + lmod] = acc[w][nt][r];
            }
        }
    }
    __syncthreads();

    // ---- per-token softmax-attend: out[t,i] = sum_j softmax_j(q_i*k_j) * v_j
    // |q*k| is O(10) for N(0,1) data, so no max-subtraction needed in fp32.
    const int i = tid & 63;
    #pragma unroll 1
    for (int it = 0; it < 32; ++it) {
        int   tok = it * 2 + (tid >> 6);
        float qi  = sq[tok][i] * LOG2E;
        float s = 0.f, o = 0.f;
        #pragma unroll 8
        for (int j = 0; j < 64; ++j) {
            float e = exp2f(qi * sk[tok][j]);   // v_exp_f32
            s += e;
            o += e * sv[tok][j];
        }
        out[(size_t)(tokBase + tok) * Z + i] = o / s;
    }
}

extern "C" void kernel_launch(void* const* d_in, const int* in_sizes, int n_in,
                              void* d_out, int out_size, void* d_ws, size_t ws_size,
                              hipStream_t stream) {
    const float* x  = (const float*)d_in[0];
    const float* Wq = (const float*)d_in[1];
    const float* Wk = (const float*)d_in[2];
    const float* Wv = (const float*)d_in[3];
    float* out        = (float*)d_out;
    unsigned int* wsw = (unsigned int*)d_ws;   // uses 3*8192*32 B = 768 KB of ws

    // 3 * 8192 chunks, one thread each
    swizzle_weights<<<(3 * CHUNKS_PER_W) / 256, 256, 0, stream>>>(Wq, Wk, Wv, wsw);
    // 16384 tokens / 64 per WG = 256 workgroups
    qkv_attn_fused<<<NTOK / TOK_PER_WG, 128, 0, stream>>>(x, wsw, out);
}
